// MultiHeadAttention_53266184405720
// MI455X (gfx1250) — matrix operand order repaired and checked by reading, not compiled, not run on hardware
//
#include <hip/hip_runtime.h>

typedef _Float16 f16;
typedef __attribute__((ext_vector_type(8)))  _Float16 v8h;
typedef __attribute__((ext_vector_type(16))) _Float16 v16h;
typedef __attribute__((ext_vector_type(8)))  float    v8f;

#define SEQ    4096
#define DH     64
#define HEADS  8
#define DMODEL 512  // DH*HEADS

union V16u { v16h v; v8h h[2]; };

// ---- WMMA fragment helpers (contiguous K order, same for A and B) ----
// A 16x32 f16: lane(l16,hf): M=l16; elems e -> K=hf*16+e
// caller passes p = base + row*ld + kBase + hf*16  (row-major, K contiguous)
__device__ __forceinline__ v16h load_a_frag(const f16* p) {
  V16u u;
  u.h[0] = *(const v8h*)(p);
  u.h[1] = *(const v8h*)(p + 8);
  return u.v;
}
// B 32x16 f16: lane(l16,hf): N=l16; elems e -> K=hf*16+e
// caller passes p = base + col*ldn + kBase + hf*16 (contiguous K per column)
__device__ __forceinline__ v16h load_b_frag(const f16* p) {
  V16u u;
  u.h[0] = *(const v8h*)(p);
  u.h[1] = *(const v8h*)(p + 8);
  return u.v;
}
__device__ __forceinline__ v8f wmma16(v16h a, v16h b, v8f c) {
  return __builtin_amdgcn_wmma_f32_16x16x32_f16(false, a, false, b, (short)0, c, false, false);
}

// ---- prep: fp32 -> fp16 copies of q,k,v ----
__global__ void prep_cvt_qkv(const float* __restrict__ q, const float* __restrict__ k,
                             const float* __restrict__ v,
                             f16* __restrict__ qh, f16* __restrict__ kh, f16* __restrict__ vh) {
  int i = blockIdx.x * blockDim.x + threadIdx.x;
  if (i < SEQ * DH) {
    qh[i] = (f16)q[i];
    kh[i] = (f16)k[i];
    vh[i] = (f16)v[i];
  }
}

// ---- prep: transpose+convert weights: w[rows][cols] -> wt[cols][rows] (f16) ----
__global__ void prep_w(const float* __restrict__ w, f16* __restrict__ wt, int rows, int cols) {
  int i = blockIdx.x * blockDim.x + threadIdx.x;
  if (i < rows * cols) {
    int n = i / rows;        // output column index
    int kk = i - n * rows;   // inner (K) index
    wt[i] = (f16)w[kk * cols + n];
  }
}

// ---- projection GEMM: [4096,64] @ Wt(colmajor f16) + bias -> scrambled head layouts ----
__global__ __launch_bounds__(32)
void proj_kernel(const f16* __restrict__ X, const f16* __restrict__ Wt,
                 const float* __restrict__ bias,
                 f16* __restrict__ dstQK,  // flat (h,s,d) identity layout
                 f16* __restrict__ dstVt,  // [h][d][s] feature-major
                 int isV) {
  int mt = blockIdx.x;             // 0..255 row tiles
  int nt = blockIdx.y;             // 0..31 col tiles
  int lane = threadIdx.x;
  int l16 = lane & 15, hf = lane >> 4;

  v8f c = {};
#pragma unroll
  for (int ko = 0; ko < 2; ++ko) {
    v16h a = load_a_frag(X + (mt * 16 + l16) * DH + ko * 32 + hf * 16);
    v16h b = load_b_frag(Wt + (nt * 16 + l16) * DH + ko * 32 + hf * 16);
    c = wmma16(a, b, c);
  }
  float bv = bias[nt * 16 + l16];
#pragma unroll
  for (int r = 0; r < 8; ++r) {
    float val = c[r] + bv;
    int row = mt * 16 + hf * 8 + r;
    int col = nt * 16 + l16;
    int flat = row * DMODEL + col;   // torch .view(h,-1,d) flat index
    if (!isV) {
      dstQK[flat] = (f16)val;
    } else {
      int h = flat >> 18;               // /(SEQ*DH)
      int s = (flat >> 6) & (SEQ - 1);  // /DH % SEQ
      int f = flat & (DH - 1);
      dstVt[(h * DH + f) * SEQ + s] = (f16)val;
    }
  }
}

// ---- flash attention: one wave per (head, 16-query tile) ----
__global__ __launch_bounds__(32)
void attn_kernel(const f16* __restrict__ Qh, const f16* __restrict__ Kh,
                 const f16* __restrict__ Vt, f16* __restrict__ ctx) {
  __shared__ __align__(16) f16 p_lds[16 * 32];

  int qt = blockIdx.x;  // 0..255
  int h  = blockIdx.y;  // 0..7
  int lane = threadIdx.x;
  int l16 = lane & 15, hf = lane >> 4;

  const f16* Q = Qh + h * SEQ * DH;
  const f16* K = Kh + h * SEQ * DH;
  const f16* V = Vt + h * DH * SEQ;

  v16h aq[2];
#pragma unroll
  for (int ko = 0; ko < 2; ++ko)
    aq[ko] = load_a_frag(Q + (qt * 16 + l16) * DH + ko * 32 + hf * 16);

  v8f zero = {};
  v8f o[4];
  o[0] = zero; o[1] = zero; o[2] = zero; o[3] = zero;
  float m_run[8], l_run[8];
#pragma unroll
  for (int r = 0; r < 8; ++r) { m_run[r] = -1e30f; l_run[r] = 0.0f; }

  // softmax runs in the base-2 log domain: v_exp_f32 is natively exp2, so we
  // fold 1/sqrt(d) * log2(e) into the score scale and avoid a mul per exp.
  const float scale2 = 0.125f * 1.44269504088896340736f;

  for (int kb = 0; kb < SEQ / 32; ++kb) {
    int key0 = kb * 32;

    // prefetch next key block's K rows and V columns (global_prefetch_b8);
    // overlaps next block's memory latency with this block's softmax VALU.
    if (kb + 1 < SEQ / 32) {
      __builtin_prefetch(K + (key0 + 32 + lane) * DH, 0, 3);
      __builtin_prefetch(V + lane * SEQ + key0 + 32, 0, 3);
      __builtin_prefetch(V + (32 + lane) * SEQ + key0 + 32, 0, 3);
    }

    // scores for 32 keys: two 16x16 WMMA outputs, K-dim 64 (2 x k32)
    v8f c0 = zero, c1 = zero;
#pragma unroll
    for (int ko = 0; ko < 2; ++ko) {
      v16h b0 = load_b_frag(K + (key0 + l16) * DH + ko * 32 + hf * 16);
      v16h b1 = load_b_frag(K + (key0 + 16 + l16) * DH + ko * 32 + hf * 16);
      c0 = wmma16(aq[ko], b0, c0);
      c1 = wmma16(aq[ko], b1, c1);
    }
    // online softmax over rows (row = hf*8+r spans the 16 lanes with same hf)
    float corr[8];
#pragma unroll
    for (int r = 0; r < 8; ++r) {
      float s0 = c0[r] * scale2, s1 = c1[r] * scale2;
      float mx = fmaxf(s0, s1);
      mx = fmaxf(mx, __shfl_xor(mx, 1, 32));
      mx = fmaxf(mx, __shfl_xor(mx, 2, 32));
      mx = fmaxf(mx, __shfl_xor(mx, 4, 32));
      mx = fmaxf(mx, __shfl_xor(mx, 8, 32));
      float m_new = fmaxf(m_run[r], mx);
      corr[r] = exp2f(m_run[r] - m_new);
      float p0 = exp2f(s0 - m_new);
      float p1 = exp2f(s1 - m_new);
      float rs = p0 + p1;
      rs += __shfl_xor(rs, 1, 32);
      rs += __shfl_xor(rs, 2, 32);
      rs += __shfl_xor(rs, 4, 32);
      rs += __shfl_xor(rs, 8, 32);
      l_run[r] = l_run[r] * corr[r] + rs;
      m_run[r] = m_new;
      int m = hf * 8 + r;
      p_lds[m * 32 + l16]      = (f16)p0;  // key cols 0..15
      p_lds[m * 32 + 16 + l16] = (f16)p1;  // key cols 16..31
    }
    __syncthreads();
    v16h pa = load_a_frag(p_lds + l16 * 32 + hf * 16);  // P as 16x32 A-fragment
    __syncthreads();

    // O(16x64) += P(16x32) @ V(32x64), with rescale correction
#pragma unroll
    for (int t = 0; t < 4; ++t) {
      v16h bv = load_b_frag(V + (t * 16 + l16) * SEQ + key0 + hf * 16);
      v8f oc = o[t];
#pragma unroll
      for (int r = 0; r < 8; ++r) oc[r] *= corr[r];
      o[t] = wmma16(pa, bv, oc);
    }
  }

  // normalize (one rcp per row instead of 32 divides) and store ctx f16
  float inv_l[8];
#pragma unroll
  for (int r = 0; r < 8; ++r) inv_l[r] = 1.0f / l_run[r];
#pragma unroll
  for (int t = 0; t < 4; ++t) {
#pragma unroll
    for (int r = 0; r < 8; ++r) {
      float val = o[t][r] * inv_l[r];
      int s = qt * 16 + hf * 8 + r;
      int f = t * 16 + l16;
      ctx[(h * SEQ + s) * DH + f] = (f16)val;
    }
  }
}

// ---- output GEMM: ctx[4096,512] @ ow[512,64] + bias -> out fp32 ----
__global__ __launch_bounds__(32)
void out_kernel(const f16* __restrict__ ctx, const f16* __restrict__ owt,
                const float* __restrict__ ob, float* __restrict__ out) {
  int mt = blockIdx.x;  // 0..255
  int nt = blockIdx.y;  // 0..3
  int lane = threadIdx.x;
  int l16 = lane & 15, hf = lane >> 4;

  v8f c = {};
#pragma unroll
  for (int ko = 0; ko < 16; ++ko) {
    v16h a = load_a_frag(ctx + (mt * 16 + l16) * DMODEL + ko * 32 + hf * 16);
    v16h b = load_b_frag(owt + (nt * 16 + l16) * DMODEL + ko * 32 + hf * 16);
    c = wmma16(a, b, c);
  }
  float bv = ob[nt * 16 + l16];
#pragma unroll
  for (int r = 0; r < 8; ++r) {
    int row = mt * 16 + hf * 8 + r;
    int col = nt * 16 + l16;
    out[row * DH + col] = c[r] + bv;
  }
}

extern "C" void kernel_launch(void* const* d_in, const int* in_sizes, int n_in,
                              void* d_out, int out_size, void* d_ws, size_t ws_size,
                              hipStream_t stream) {
  (void)in_sizes; (void)n_in; (void)out_size; (void)ws_size;
  const float* q    = (const float*)d_in[0];
  const float* k    = (const float*)d_in[1];
  const float* v    = (const float*)d_in[2];
  const float* qw_w = (const float*)d_in[3];
  const float* qw_b = (const float*)d_in[4];
  const float* kw_w = (const float*)d_in[5];
  const float* kw_b = (const float*)d_in[6];
  const float* vw_w = (const float*)d_in[7];
  const float* vw_b = (const float*)d_in[8];
  const float* ow_w = (const float*)d_in[9];
  const float* ow_b = (const float*)d_in[10];
  float* out = (float*)d_out;

  char* ws = (char*)d_ws;
  size_t off = 0;
  auto carve = [&](size_t bytes) {
    void* p = ws + off;
    off += (bytes + 255) & ~(size_t)255;
    return p;
  };
  f16* Xq  = (f16*)carve(SEQ * DH * sizeof(f16));
  f16* Xk  = (f16*)carve(SEQ * DH * sizeof(f16));
  f16* Xv  = (f16*)carve(SEQ * DH * sizeof(f16));
  f16* Wqt = (f16*)carve(DH * DMODEL * sizeof(f16));
  f16* Wkt = (f16*)carve(DH * DMODEL * sizeof(f16));
  f16* Wvt = (f16*)carve(DH * DMODEL * sizeof(f16));
  f16* OWt = (f16*)carve(DMODEL * DH * sizeof(f16));
  f16* Qh  = (f16*)carve((size_t)SEQ * DMODEL * sizeof(f16));
  f16* Kh  = (f16*)carve((size_t)SEQ * DMODEL * sizeof(f16));
  f16* Vt  = (f16*)carve((size_t)SEQ * DMODEL * sizeof(f16));
  f16* CTX = (f16*)carve((size_t)SEQ * DMODEL * sizeof(f16));

  prep_cvt_qkv<<<(SEQ * DH + 255) / 256, 256, 0, stream>>>(q, k, v, Xq, Xk, Xv);
  prep_w<<<(DH * DMODEL + 255) / 256, 256, 0, stream>>>(qw_w, Wqt, DH, DMODEL);
  prep_w<<<(DH * DMODEL + 255) / 256, 256, 0, stream>>>(kw_w, Wkt, DH, DMODEL);
  prep_w<<<(DH * DMODEL + 255) / 256, 256, 0, stream>>>(vw_w, Wvt, DH, DMODEL);
  prep_w<<<(DH * DMODEL + 255) / 256, 256, 0, stream>>>(ow_w, OWt, DMODEL, DH);

  dim3 pg(SEQ / 16, DMODEL / 16);
  proj_kernel<<<pg, 32, 0, stream>>>(Xq, Wqt, qw_b, Qh, Vt, 0);
  proj_kernel<<<pg, 32, 0, stream>>>(Xk, Wkt, kw_b, Kh, Vt, 0);
  proj_kernel<<<pg, 32, 0, stream>>>(Xv, Wvt, vw_b, Qh, Vt, 1);

  attn_kernel<<<dim3(SEQ / 16, HEADS), 32, 0, stream>>>(Qh, Kh, Vt, CTX);

  out_kernel<<<dim3(SEQ / 16, DH / 16), 32, 0, stream>>>(CTX, OWt, ow_b, out);
}
